// MultiHeadAttentionLayer_46961172414541
// MI455X (gfx1250) — matrix operand order repaired and checked by reading, not compiled, not run on hardware
//
#include <hip/hip_runtime.h>
#include <math.h>

#define N_NODES   10000
#define N_EDGES   640000
#define IN_DIM    128
#define NUM_HEADS 8
#define OUT_DIM   16
#define HD        128   // NUM_HEADS * OUT_DIM
#define W_ELEMS   (HD * IN_DIM)   // 16384 per weight matrix

typedef __attribute__((ext_vector_type(8)))  float     v8f;
typedef __attribute__((ext_vector_type(16))) _Float16  v16h;

#define AST 136   // halfs per LDS row (pad: 68 dwords, 68 % 64 = 4 -> conflict-free)

// ---------------------------------------------------------------------------
// Stage 16 rows x 128 f32 of `g` into LDS as split f16 hi/lo arrays.
// 256 threads, each converts 8 consecutive floats of one row.
// ---------------------------------------------------------------------------
__device__ __forceinline__ void stage_split_rows(_Float16* AhL, _Float16* AlL,
                                                 const float* __restrict__ g,
                                                 long long rowBase, int tid) {
    const int r = tid >> 4;          // 0..15
    const int c = (tid & 15) * 8;    // 0,8,...,120
    const float4* gp = (const float4*)(g + (rowBase + r) * (long long)IN_DIM + c);
    float4 v0 = gp[0];
    float4 v1 = gp[1];
    float x[8] = {v0.x, v0.y, v0.z, v0.w, v1.x, v1.y, v1.z, v1.w};
    union { _Float16 h[8]; float4 f4; } Hi, Lo;
#pragma unroll
    for (int i = 0; i < 8; ++i) {
        _Float16 hv = (_Float16)x[i];
        Hi.h[i] = hv;
        Lo.h[i] = (_Float16)(x[i] - (float)hv);
    }
    *(float4*)(AhL + r * AST + c) = Hi.f4;   // 16B aligned (272r + 16t)
    *(float4*)(AlL + r * AST + c) = Lo.f4;
}

// ---------------------------------------------------------------------------
// One 16x16 f32 tile of  C = A(16x128) * W^T(cols h16..h16+15) + bias
// via split-f16 WMMA:  C += Ah*Bh + Ah*Bl + Al*Bh  per K=32 block.
// A layout (16-bit A 16x32): lane&15 = M; VGPR v holds K = kb + 16*(v>>2)
//   + 8*(lane>>4) + 2*(v&3) (+1 in high half of dword).
// B layout (32x16): lane&15 = N; VGPR v holds K = kb + 16*(v>>2)
//   + 8*(lane>>4) + 2*(v&3) (+1).
// ---------------------------------------------------------------------------
__device__ __forceinline__ v8f gemm_tile_split(const _Float16* AhL,
                                               const _Float16* AlL,
                                               const _Float16* __restrict__ Wh,
                                               const _Float16* __restrict__ Wl,
                                               const float* __restrict__ bias,
                                               int h16, int lane) {
    const int n   = lane & 15;
    const int hlf = lane >> 4;
    const int m   = lane & 15;

    v8f c;
    const float bval = bias[h16 + n];
#pragma unroll
    for (int j = 0; j < 8; ++j) c[j] = bval;

    const _Float16* ahrow = AhL + m * AST;
    const _Float16* alrow = AlL + m * AST;
    const _Float16* bhrow = Wh + (size_t)(h16 + n) * IN_DIM + hlf * 8;
    const _Float16* blrow = Wl + (size_t)(h16 + n) * IN_DIM + hlf * 8;

#pragma unroll
    for (int kb = 0; kb < IN_DIM; kb += 32) {
        union { v16h v; unsigned u[8]; } Ah, Al, Bh, Bl;
#pragma unroll
        for (int v = 0; v < 8; ++v) {
            const int ka = kb + ((v >> 2) << 4) + (hlf << 3) + ((v & 3) << 1);
            Ah.u[v] = *(const unsigned*)(ahrow + ka);
            Al.u[v] = *(const unsigned*)(alrow + ka);
            const int kbv = kb + ((v >> 2) << 4) + ((v & 3) << 1);
            Bh.u[v] = *(const unsigned*)(bhrow + kbv);
            Bl.u[v] = *(const unsigned*)(blrow + kbv);
        }
        c = __builtin_amdgcn_wmma_f32_16x16x32_f16(false, Ah.v, false, Bh.v,
                                                   (short)0, c, false, false);
        c = __builtin_amdgcn_wmma_f32_16x16x32_f16(false, Ah.v, false, Bl.v,
                                                   (short)0, c, false, false);
        c = __builtin_amdgcn_wmma_f32_16x16x32_f16(false, Al.v, false, Bh.v,
                                                   (short)0, c, false, false);
    }
    return c;
}

// ---------------- Kernel 0: split weights into f16 hi/lo -------------------
__global__ __launch_bounds__(256) void gt_prep_w_kernel(
    const float* __restrict__ Wq, const float* __restrict__ Wk,
    const float* __restrict__ Wv, const float* __restrict__ We,
    _Float16* __restrict__ Wh, _Float16* __restrict__ Wl)
{
    const int mat = blockIdx.y;   // 0=q,1=k,2=v,3=e
    const float* W = (mat == 0) ? Wq : (mat == 1) ? Wk : (mat == 2) ? Wv : We;
    const int i = blockIdx.x * blockDim.x + threadIdx.x;   // 0..16383
    const float x = W[i];
    const _Float16 hv = (_Float16)x;
    Wh[mat * W_ELEMS + i] = hv;
    Wl[mat * W_ELEMS + i] = (_Float16)(x - (float)hv);
}

// ---------------- Kernel 1: node projections Q,K,V = h @ W.T + b ----------
__global__ __launch_bounds__(256) void gt_qkv_kernel(
    const float* __restrict__ h,
    const _Float16* __restrict__ Wh, const _Float16* __restrict__ Wl,
    const float* __restrict__ bq, const float* __restrict__ bk,
    const float* __restrict__ bv,
    float* __restrict__ Q, float* __restrict__ K, float* __restrict__ V)
{
    __shared__ _Float16 AhL[16 * AST];
    __shared__ _Float16 AlL[16 * AST];
    const int tid = threadIdx.x;
    const long long tile = blockIdx.x;            // 625 tiles of 16 nodes

    stage_split_rows(AhL, AlL, h, tile * 16, tid);
    __syncthreads();

    const int mat = blockIdx.y;                   // 0=Q,1=K,2=V
    const float* b = (mat == 0) ? bq : (mat == 1) ? bk : bv;
    float* out     = (mat == 0) ? Q  : (mat == 1) ? K  : V;
    const _Float16* wh = Wh + (size_t)mat * W_ELEMS;
    const _Float16* wl = Wl + (size_t)mat * W_ELEMS;

    const int wave = tid >> 5;                    // head index 0..7
    const int lane = tid & 31;
    const int h16  = wave * OUT_DIM;

    v8f c = gemm_tile_split(AhL, AlL, wh, wl, b, h16, lane);

    const int n = lane & 15, hlf = lane >> 4;
#pragma unroll
    for (int j = 0; j < 8; ++j) {
        long long row = tile * 16 + j + hlf * 8;
        out[row * HD + h16 + n] = c[j];
    }
}

// ---------------- Kernel 2: fused edge projection + attention scatter -----
__global__ __launch_bounds__(256) void gt_edge_kernel(
    const float* __restrict__ e,
    const int*   __restrict__ src, const int* __restrict__ dst,
    const _Float16* __restrict__ Weh, const _Float16* __restrict__ Wel,
    const float* __restrict__ be,
    const float* __restrict__ Q, const float* __restrict__ K,
    const float* __restrict__ V,
    float* __restrict__ wV, float* __restrict__ z)
{
    __shared__ _Float16 AhL[16 * AST];
    __shared__ _Float16 AlL[16 * AST];
    __shared__ int lsrc[16], ldst[16];
    const int tid = threadIdx.x;
    const long long tile = blockIdx.x;            // 40000 tiles of 16 edges

    stage_split_rows(AhL, AlL, e, tile * 16, tid);
    if (tid < 16) {
        lsrc[tid] = src[tile * 16 + tid];
        ldst[tid] = dst[tile * 16 + tid];
    }
    __syncthreads();

    const int wave = tid >> 5;                    // head index 0..7
    const int lane = tid & 31;
    const int h16  = wave * OUT_DIM;

    // E_e tile for 16 edges x 16 dims of this head (C layout)
    v8f c = gemm_tile_split(AhL, AlL, Weh, Wel, be, h16, lane);

    const int n   = lane & 15;                    // per-head dim d
    const int hlf = lane >> 4;
    const int col = h16 + n;

#pragma unroll
    for (int j = 0; j < 8; ++j) {
        const int m  = j + hlf * 8;               // edge index within tile
        const int si = lsrc[m];
        const int di = ldst[m];

        const float kq = K[(long long)si * HD + col] * Q[(long long)di * HD + col];
        float t = c[j] * kq * 0.25f;              // / sqrt(OUT_DIM)

        // reduce over d across the 16 lanes of this half-wave
        t += __shfl_xor(t, 1, 16);
        t += __shfl_xor(t, 2, 16);
        t += __shfl_xor(t, 4, 16);
        t += __shfl_xor(t, 8, 16);

        const float s = expf(fminf(fmaxf(t, -5.0f), 5.0f));

        const float val = V[(long long)si * HD + col] * s;
        atomicAdd(&wV[(long long)di * HD + col], val);
        if (n == 0) atomicAdd(&z[(long long)di * NUM_HEADS + wave], s);
    }
}

// ---------------- Kernel 3: normalize -------------------------------------
__global__ __launch_bounds__(256) void gt_final_kernel(
    const float* __restrict__ wV, const float* __restrict__ z,
    float* __restrict__ out)
{
    const int idx = blockIdx.x * blockDim.x + threadIdx.x;
    if (idx >= N_NODES * HD) return;
    const int nrow = idx >> 7;        // / HD
    const int col  = idx & (HD - 1);
    const int head = col >> 4;        // / OUT_DIM
    out[idx] = wV[idx] / (z[nrow * NUM_HEADS + head] + 1e-6f);
}

extern "C" void kernel_launch(void* const* d_in, const int* in_sizes, int n_in,
                              void* d_out, int out_size, void* d_ws, size_t ws_size,
                              hipStream_t stream) {
    const float* h   = (const float*)d_in[0];
    const float* e   = (const float*)d_in[1];
    const int*   src = (const int*)d_in[2];
    const int*   dst = (const int*)d_in[3];
    const float* Wq  = (const float*)d_in[4];
    const float* bq  = (const float*)d_in[5];
    const float* Wk  = (const float*)d_in[6];
    const float* bk  = (const float*)d_in[7];
    const float* We  = (const float*)d_in[8];
    const float* be  = (const float*)d_in[9];
    const float* Wv  = (const float*)d_in[10];
    const float* bv  = (const float*)d_in[11];
    float* out = (float*)d_out;

    // workspace layout (floats): Q | K | V | wV | z | Wh(f16) | Wl(f16)
    const size_t NODE_F = (size_t)N_NODES * HD;        // 1,280,000
    float* Q  = (float*)d_ws;
    float* K  = Q  + NODE_F;
    float* V  = K  + NODE_F;
    float* wV = V  + NODE_F;
    float* z  = wV + NODE_F;
    _Float16* Wh = (_Float16*)(z + (size_t)N_NODES * NUM_HEADS);
    _Float16* Wl = Wh + (size_t)4 * W_ELEMS;

    // zero accumulators every call (graph replays do not re-initialize ws)
    hipMemsetAsync(wV, 0,
                   (NODE_F + (size_t)N_NODES * NUM_HEADS) * sizeof(float),
                   stream);

    // split weights to f16 hi/lo (order in Wh/Wl: q,k,v,e)
    dim3 gp(W_ELEMS / 256, 4);
    gt_prep_w_kernel<<<gp, 256, 0, stream>>>(Wq, Wk, Wv, We, Wh, Wl);

    dim3 g1(N_NODES / 16, 3);
    gt_qkv_kernel<<<g1, 256, 0, stream>>>(h, Wh, Wl, bq, bk, bv, Q, K, V);

    const _Float16* Weh = Wh + (size_t)3 * W_ELEMS;
    const _Float16* Wel = Wl + (size_t)3 * W_ELEMS;
    gt_edge_kernel<<<N_EDGES / 16, 256, 0, stream>>>(e, src, dst, Weh, Wel, be,
                                                     Q, K, V, wV, z);

    const int tot = N_NODES * HD;
    gt_final_kernel<<<(tot + 255) / 256, 256, 0, stream>>>(wV, z, out);
}
